// Attention_2181843386441
// MI455X (gfx1250) — compile-verified, matrix operand order checked by reading, not run on hardware
//
#include <hip/hip_runtime.h>

// ---------------------------------------------------------------------------
// BEiT-style attention for MI455X (gfx1250, wave32, WMMA).
// All three matmul stages run on v_wmma_f32_16x16x32_bf16 with f32 accum.
// Fragments are loaded as 2x b128 per 16x32 tile (layouts per ISA 7.12.2).
// proj_kernel's bf16 A-tiles use GLOBAL_LOAD_ASYNC_TO_LDS_B128 (ASYNCcnt)
// when the builtin is available.
// Workspace layout (bytes): qws | kws | vws (bf16 [B,H,N,64] each)
//                           | rel_bias f32 [H,N,N] | attn_out bf16 [B*N,768]
// Total ~80 MB.
// ---------------------------------------------------------------------------

typedef __attribute__((ext_vector_type(16))) __bf16 v16bf;
typedef __attribute__((ext_vector_type(8)))  float  v8f;
typedef __attribute__((ext_vector_type(4)))  int    v4i;

#define BATCH 64
#define NTOK  197
#define DIMC  768
#define NH    12
#define HD    64
#define MTOT  (BATCH * NTOK)          // 12608 rows of the token matrix
#define LSTR  40                      // LDS tile stride (80B: 16B-aligned rows)

#if defined(__has_builtin)
#if __has_builtin(__builtin_amdgcn_global_load_async_to_lds_b128) && \
    __has_builtin(__builtin_amdgcn_s_wait_asynccnt)
#define HAVE_ASYNC_LDS 1
#endif
#endif

#ifdef HAVE_ASYNC_LDS
typedef __attribute__((address_space(1))) v4i* gv4i_p;   // global
typedef __attribute__((address_space(3))) v4i* lv4i_p;   // LDS
#endif

__device__ inline v8f zero8() {
  v8f z;
#pragma unroll
  for (int i = 0; i < 8; ++i) z[i] = 0.0f;
  return z;
}

__device__ inline v8f bwmma(v16bf a, v16bf b, v8f c) {
  // (neg_a, A, neg_b, B, c_mod, C, reuse_a, reuse_b)
  return __builtin_amdgcn_wmma_f32_16x16x32_bf16(false, a, false, b,
                                                 (short)0, c, false, false);
}

// 16x32 bf16 fragment (A-layout; B uses the same per-lane layout with N in
// place of M).  rowPtr already points at this lane's row (lane&15).
// ISA 7.12.2: VGPR v, half h, group g: K = 16*(v>=4) + 8*g + 2*(v&3) + h
// => per lane: halves [0..7] = rowPtr[kB+kg .. +7], [8..15] = rowPtr[kB+16+kg ..]
// i.e. two contiguous 16B runs -> two b128 loads.  Requires 16B alignment of
// rowPtr+kB (kg is 0 or 8 elements = 0/16B).
__device__ inline v16bf frag_row(const __bf16* rowPtr, int kB, int lane) {
  const int kg = (lane & 16) ? 8 : 0;
  union { uint4 u[2]; v16bf f; } u;
  u.u[0] = *(const uint4*)(rowPtr + kB + kg);
  u.u[1] = *(const uint4*)(rowPtr + kB + kg + 16);
  return u.f;
}

// ---------------------------------------------------------------------------
// Kernel 0: rel_bias[h][q][k] = rel_table[rel_index[q][k]][h]
// ---------------------------------------------------------------------------
__global__ __launch_bounds__(256) void bias_kernel(
    const float* __restrict__ rel_table, const int* __restrict__ rel_index,
    float* __restrict__ bias) {
  int i = blockIdx.x * 256 + threadIdx.x;
  const int tot = NH * NTOK * NTOK;
  if (i >= tot) return;
  int h  = i / (NTOK * NTOK);
  int qk = i - h * (NTOK * NTOK);
  bias[i] = rel_table[rel_index[qk] * NH + h];
}

// ---------------------------------------------------------------------------
// Kernel 1: qkv = x @ Wqkv^T + bias ; split into scaled-Q / K / V, bf16,
//           layout [B,H,N,64].  Block tile 128(M) x 64(N), 8 waves.
// ---------------------------------------------------------------------------
__global__ __launch_bounds__(256) void qkv_kernel(
    const float* __restrict__ x, const float* __restrict__ Wqkv,
    const float* __restrict__ qbias, const float* __restrict__ vbias,
    __bf16* __restrict__ qws, __bf16* __restrict__ kws,
    __bf16* __restrict__ vws) {
  __shared__ __attribute__((aligned(16))) __bf16 As[128 * LSTR];
  __shared__ __attribute__((aligned(16))) __bf16 Bs[64 * LSTR];

  const int rb   = blockIdx.y * 128;
  const int jb   = blockIdx.x * 64;
  const int lane = threadIdx.x & 31;
  const int wave = threadIdx.x >> 5;

  v8f acc[4];
#pragma unroll
  for (int nt = 0; nt < 4; ++nt) acc[nt] = zero8();

  for (int kt = 0; kt < DIMC / 32; ++kt) {
    const int kb = kt * 32;
    __syncthreads();
    {  // stage A: 128 x 32 fp32 -> bf16
      int r = threadIdx.x >> 1, seg = threadIdx.x & 1;
      int rg = rb + r;
      const float4* p = (const float4*)(x + (size_t)rg * DIMC + kb + seg * 16);
      if (kt + 1 < DIMC / 32 && rg < MTOT)
        __builtin_prefetch(x + (size_t)rg * DIMC + kb + 32 + seg * 16, 0, 1);
      __bf16* dst = As + r * LSTR + seg * 16;
#pragma unroll
      for (int i = 0; i < 4; ++i) {
        float4 t = (rg < MTOT) ? p[i] : make_float4(0.f, 0.f, 0.f, 0.f);
        dst[4 * i + 0] = (__bf16)t.x; dst[4 * i + 1] = (__bf16)t.y;
        dst[4 * i + 2] = (__bf16)t.z; dst[4 * i + 3] = (__bf16)t.w;
      }
    }
    {  // stage B: 64 x 32 fp32 -> bf16  (Wqkv rows are K-contiguous)
      int r = threadIdx.x >> 2, seg = threadIdx.x & 3;
      const float4* p =
          (const float4*)(Wqkv + (size_t)(jb + r) * DIMC + kb + seg * 8);
      __bf16* dst = Bs + r * LSTR + seg * 8;
#pragma unroll
      for (int i = 0; i < 2; ++i) {
        float4 t = p[i];
        dst[4 * i + 0] = (__bf16)t.x; dst[4 * i + 1] = (__bf16)t.y;
        dst[4 * i + 2] = (__bf16)t.z; dst[4 * i + 3] = (__bf16)t.w;
      }
    }
    __syncthreads();
    const __bf16* arow = As + (wave * 16 + (lane & 15)) * LSTR;
    v16bf af = frag_row(arow, 0, lane);
#pragma unroll
    for (int nt = 0; nt < 4; ++nt) {
      v16bf bf = frag_row(Bs + (nt * 16 + (lane & 15)) * LSTR, 0, lane);
      acc[nt] = bwmma(af, bf, acc[nt]);
    }
  }

  // epilogue: route column j = [which*768 + h*64 + d] into q/k/v ws
  const int nloc = lane & 15;
  const int mtop = (lane >> 4) << 3;
#pragma unroll
  for (int nt = 0; nt < 4; ++nt) {
#pragma unroll
    for (int v = 0; v < 8; ++v) {
      int r = rb + wave * 16 + mtop + v;
      if (r >= MTOT) continue;
      int j  = jb + nt * 16 + nloc;
      int wh = j / DIMC;            // 0=q 1=k 2=v
      int jj = j - wh * DIMC;
      int h = jj >> 6, d = jj & 63;
      int b = r / NTOK, t = r - b * NTOK;
      size_t off = (((size_t)b * NH + h) * NTOK + t) * HD + d;
      float val = acc[nt][v];
      if (wh == 0)      qws[off] = (__bf16)((val + qbias[jj]) * 0.125f);
      else if (wh == 1) kws[off] = (__bf16)val;
      else              vws[off] = (__bf16)(val + vbias[jj]);
    }
  }
}

// ---------------------------------------------------------------------------
// Kernel 2: per (b,h) attention.  128 threads = 4 waves; each wave owns
// 16-query tiles (13 tiles cover N=197 padded to 208). Scores for all 13 key
// tiles live in v8f accumulators; softmax fp32; P goes through per-wave LDS
// to re-layout C->A for the P@V WMMA chain.  V is transposed in LDS.
// ---------------------------------------------------------------------------
__global__ __launch_bounds__(128) void attn_kernel(
    const __bf16* __restrict__ qws, const __bf16* __restrict__ kws,
    const __bf16* __restrict__ vws, const float* __restrict__ bias,
    __bf16* __restrict__ aout) {
  __shared__ __attribute__((aligned(16))) __bf16 Vt[64 * 224];
  __shared__ __attribute__((aligned(16))) __bf16 Pbuf[4 * 16 * 224];

  const int bh = blockIdx.x;
  const int b = bh / NH, h = bh - b * NH;
  const __bf16* Qb = qws + (size_t)bh * NTOK * HD;
  const __bf16* Kb = kws + (size_t)bh * NTOK * HD;
  const __bf16* Vb = vws + (size_t)bh * NTOK * HD;

  for (int idx = threadIdx.x; idx < 224 * 64; idx += 128) {
    int r = idx >> 6, d = idx & 63;
    Vt[d * 224 + r] = (r < NTOK) ? Vb[(size_t)r * HD + d] : (__bf16)0.f;
  }
  __syncthreads();

  const int lane = threadIdx.x & 31;
  const int wave = threadIdx.x >> 5;
  const int nloc = lane & 15;
  const int mtop = (lane >> 4) << 3;
  __bf16* Pw = Pbuf + wave * 16 * 224;

  // zero the K-padding columns 208..223 once (stay zero thereafter)
  for (int i = lane; i < 16 * 16; i += 32)
    Pw[(i >> 4) * 224 + 208 + (i & 15)] = (__bf16)0.f;

  for (int qt = wave; qt < 13; qt += 4) {
    const int qb0 = qt * 16;
    int qr = qb0 + nloc; if (qr > NTOK - 1) qr = NTOK - 1;   // clamp pad rows
    const __bf16* qrow = Qb + (size_t)qr * HD;
    v16bf qf0 = frag_row(qrow, 0, lane);
    v16bf qf1 = frag_row(qrow, 32, lane);

    v8f sc[13];
#pragma unroll
    for (int t = 0; t < 13; ++t) {
      sc[t] = zero8();
      int kr = t * 16 + nloc; if (kr > NTOK - 1) kr = NTOK - 1;
      const __bf16* krow = Kb + (size_t)kr * HD;   // B-frag: lane = key row
      v16bf kf0 = frag_row(krow, 0, lane);
      v16bf kf1 = frag_row(krow, 32, lane);
      sc[t] = bwmma(qf0, kf0, sc[t]);
      sc[t] = bwmma(qf1, kf1, sc[t]);
    }

    // + rel_bias, key mask, row max
    float rmax[8];
#pragma unroll
    for (int v = 0; v < 8; ++v) rmax[v] = -1e30f;
#pragma unroll
    for (int t = 0; t < 13; ++t) {
      const int kg = t * 16 + nloc;
#pragma unroll
      for (int v = 0; v < 8; ++v) {
        float s = sc[t][v];
        int q = qb0 + mtop + v; if (q > NTOK - 1) q = NTOK - 1;
        s = (kg < NTOK) ? s + bias[((size_t)h * NTOK + q) * NTOK + kg] : -1e30f;
        sc[t][v] = s;
        rmax[v] = fmaxf(rmax[v], s);
      }
    }
#pragma unroll
    for (int m = 1; m < 16; m <<= 1)
#pragma unroll
      for (int v = 0; v < 8; ++v)
        rmax[v] = fmaxf(rmax[v], __shfl_xor(rmax[v], m, 32));

    float rsum[8];
#pragma unroll
    for (int v = 0; v < 8; ++v) rsum[v] = 0.f;
#pragma unroll
    for (int t = 0; t < 13; ++t)
#pragma unroll
      for (int v = 0; v < 8; ++v) {
        float e = __expf(sc[t][v] - rmax[v]);
        sc[t][v] = e;
        rsum[v] += e;
      }
#pragma unroll
    for (int m = 1; m < 16; m <<= 1)
#pragma unroll
      for (int v = 0; v < 8; ++v) rsum[v] += __shfl_xor(rsum[v], m, 32);

    float rinv[8];
#pragma unroll
    for (int v = 0; v < 8; ++v) rinv[v] = 1.0f / rsum[v];

    // store normalized P into LDS in row-major A layout
#pragma unroll
    for (int t = 0; t < 13; ++t)
#pragma unroll
      for (int v = 0; v < 8; ++v)
        Pw[(mtop + v) * 224 + t * 16 + nloc] = (__bf16)(sc[t][v] * rinv[v]);

    // O = P @ V  (M=16, N=64, K=224 in 7 steps of 32)
    v8f o[4];
#pragma unroll
    for (int nt = 0; nt < 4; ++nt) o[nt] = zero8();
#pragma unroll
    for (int ks = 0; ks < 7; ++ks) {
      v16bf pf = frag_row(Pw + nloc * 224, ks * 32, lane);
#pragma unroll
      for (int nt = 0; nt < 4; ++nt) {
        v16bf vf = frag_row(Vt + (nt * 16 + nloc) * 224, ks * 32, lane);
        o[nt] = bwmma(pf, vf, o[nt]);
      }
    }
#pragma unroll
    for (int nt = 0; nt < 4; ++nt)
#pragma unroll
      for (int v = 0; v < 8; ++v) {
        int q = qb0 + mtop + v;
        if (q < NTOK)
          aout[((size_t)b * NTOK + q) * DIMC + h * HD + nt * 16 + nloc] =
              (__bf16)o[nt][v];
      }
  }
}

// ---------------------------------------------------------------------------
// Kernel 3: out = attn_out @ Wproj^T + bproj   (fp32 out)
// A-tiles are bf16 already: copy global->LDS with the CDNA5 async path
// (GLOBAL_LOAD_ASYNC_TO_LDS_B128 + s_wait_asynccnt) when available.
// ---------------------------------------------------------------------------
__global__ __launch_bounds__(256) void proj_kernel(
    const __bf16* __restrict__ aout, const float* __restrict__ Wproj,
    const float* __restrict__ bproj, float* __restrict__ out) {
  __shared__ __attribute__((aligned(16))) __bf16 As[128 * LSTR];
  __shared__ __attribute__((aligned(16))) __bf16 Bs[64 * LSTR];

  const int rb   = blockIdx.y * 128;
  const int jb   = blockIdx.x * 64;
  const int lane = threadIdx.x & 31;
  const int wave = threadIdx.x >> 5;

  v8f acc[4];
#pragma unroll
  for (int nt = 0; nt < 4; ++nt) acc[nt] = zero8();

  for (int kt = 0; kt < DIMC / 32; ++kt) {
    const int kb = kt * 32;
    __syncthreads();
    {  // stage A (already bf16): 128 x 32, two 16B chunks per thread
      int r = threadIdx.x >> 1, seg = threadIdx.x & 1;
      int rg = rb + r;
      if (rg >= MTOT) rg = MTOT - 1;  // clamp: dup valid row, outputs masked
      const __bf16* g = aout + (size_t)rg * DIMC + kb + seg * 16;
      __bf16* l = As + r * LSTR + seg * 16;
#ifdef HAVE_ASYNC_LDS
      __builtin_amdgcn_global_load_async_to_lds_b128((gv4i_p)g, (lv4i_p)l, 0,
                                                     0);
      __builtin_amdgcn_global_load_async_to_lds_b128((gv4i_p)(g + 8),
                                                     (lv4i_p)(l + 8), 0, 0);
#else
      const uint4* p = (const uint4*)g;
      unsigned int* d32 = (unsigned int*)l;
#pragma unroll
      for (int i = 0; i < 2; ++i) {
        uint4 t = p[i];
        d32[4 * i + 0] = t.x; d32[4 * i + 1] = t.y;
        d32[4 * i + 2] = t.z; d32[4 * i + 3] = t.w;
      }
#endif
    }
    {  // stage B: 64 x 32 fp32 -> bf16
      int r = threadIdx.x >> 2, seg = threadIdx.x & 3;
      const float4* p =
          (const float4*)(Wproj + (size_t)(jb + r) * DIMC + kb + seg * 8);
      __bf16* dst = Bs + r * LSTR + seg * 8;
#pragma unroll
      for (int i = 0; i < 2; ++i) {
        float4 t = p[i];
        dst[4 * i + 0] = (__bf16)t.x; dst[4 * i + 1] = (__bf16)t.y;
        dst[4 * i + 2] = (__bf16)t.z; dst[4 * i + 3] = (__bf16)t.w;
      }
    }
#ifdef HAVE_ASYNC_LDS
    __builtin_amdgcn_s_wait_asynccnt(0);
#endif
    __syncthreads();
    const __bf16* arow = As + (wave * 16 + (lane & 15)) * LSTR;
    v16bf af = frag_row(arow, 0, lane);
#pragma unroll
    for (int nt = 0; nt < 4; ++nt) {
      v16bf bf = frag_row(Bs + (nt * 16 + (lane & 15)) * LSTR, 0, lane);
      acc[nt] = bwmma(af, bf, acc[nt]);
    }
  }

  const int nloc = lane & 15;
  const int mtop = (lane >> 4) << 3;
#pragma unroll
  for (int nt = 0; nt < 4; ++nt)
#pragma unroll
    for (int v = 0; v < 8; ++v) {
      int r = rb + wave * 16 + mtop + v;
      if (r >= MTOT) continue;
      int j = jb + nt * 16 + nloc;
      out[(size_t)r * DIMC + j] = acc[nt][v] + bproj[j];
    }
}

// ---------------------------------------------------------------------------
extern "C" void kernel_launch(void* const* d_in, const int* in_sizes, int n_in,
                              void* d_out, int out_size, void* d_ws,
                              size_t ws_size, hipStream_t stream) {
  const float* x         = (const float*)d_in[0];
  const float* Wqkv      = (const float*)d_in[1];
  const float* q_bias    = (const float*)d_in[2];
  const float* v_bias    = (const float*)d_in[3];
  const float* rel_table = (const float*)d_in[4];
  const float* Wproj     = (const float*)d_in[5];
  const float* bproj     = (const float*)d_in[6];
  const int*   rel_index = (const int*)d_in[7];
  float* out = (float*)d_out;

  char* ws = (char*)d_ws;
  const size_t S1 = (size_t)BATCH * NH * NTOK * HD * sizeof(__bf16);  // 19.4MB
  const size_t BIAS_BYTES =
      ((((size_t)NH * NTOK * NTOK * sizeof(float)) + 255) / 256) * 256;
  __bf16* qws  = (__bf16*)(ws);
  __bf16* kws  = (__bf16*)(ws + S1);
  __bf16* vws  = (__bf16*)(ws + 2 * S1);
  float*  bias = (float*)(ws + 3 * S1);
  __bf16* aout = (__bf16*)(ws + 3 * S1 + BIAS_BYTES);

  {  // rel_bias[H,N,N]
    int tot = NH * NTOK * NTOK;
    bias_kernel<<<(tot + 255) / 256, 256, 0, stream>>>(rel_table, rel_index,
                                                       bias);
  }
  {  // QKV projection
    dim3 grid((3 * DIMC) / 64, (MTOT + 127) / 128);
    qkv_kernel<<<grid, 256, 0, stream>>>(x, Wqkv, q_bias, v_bias, qws, kws,
                                         vws);
  }
  {  // attention per (b,h)
    attn_kernel<<<BATCH * NH, 128, 0, stream>>>(qws, kws, vws, bias, aout);
  }
  {  // output projection
    dim3 grid(DIMC / 64, (MTOT + 127) / 128);
    proj_kernel<<<grid, 256, 0, stream>>>(aout, Wproj, bproj, out);
  }
}
